// GraphBiasedMultiHeadAttention_56530359550887
// MI455X (gfx1250) — matrix operand order repaired and checked by reading, not compiled, not run on hardware
//
#include <hip/hip_runtime.h>
#include <hip/hip_bf16.h>
#include <math.h>

#define NSEQ   2048
#define HID    128
#define NHEADS 8
#define HDIM   16
#define QKSCALE 0.25f   // HEAD_DIM^-0.5

typedef __attribute__((ext_vector_type(16))) _Float16 v16h;
typedef __attribute__((ext_vector_type(8)))  _Float16 v8h;
typedef __attribute__((ext_vector_type(8)))  float    v8f;

// exact pointee type expected by the async builtin (from hipcc diagnostic):
// 'int __attribute__((vector_size(16)))' pointer, global/local address spaces
typedef int v4i __attribute__((vector_size(16)));
typedef __attribute__((address_space(1))) v4i as1_v4i;
typedef __attribute__((address_space(3))) v4i as3_v4i;

// ---- CDNA5 async global->LDS copy (ASYNCcnt path), with safe fallback ------
#if defined(__has_builtin)
# if __has_builtin(__builtin_amdgcn_global_load_async_to_lds_b128)
#  define HAVE_ASYNC_LDS 1
# else
#  define HAVE_ASYNC_LDS 0
# endif
#else
# define HAVE_ASYNC_LDS 0
#endif

static __device__ __forceinline__ void async_copy16(const float* g, float* l) {
#if HAVE_ASYNC_LDS
  __builtin_amdgcn_global_load_async_to_lds_b128((as1_v4i*)g, (as3_v4i*)l,
                                                 0 /*imm offset*/, 0 /*cpol*/);
#else
  *(float4*)l = *(const float4*)g;   // coalesced fallback through VGPRs
#endif
}

template <int N>
static __device__ __forceinline__ void wait_asynccnt() {
#if HAVE_ASYNC_LDS
# if __has_builtin(__builtin_amdgcn_s_wait_asynccnt)
  __builtin_amdgcn_s_wait_asynccnt(N);
# else
  if constexpr (N == 0) { __asm__ volatile("s_wait_asynccnt 0x0" ::: "memory"); }
  else                  { __asm__ volatile("s_wait_asynccnt 0x4" ::: "memory"); }
# endif
#endif
}

static __device__ inline v8f wmma16(v16h a, v16h b, v8f c) {
  // D = A(16x32 f16) * B(32x16 f16) + C(16x16 f32)
  return __builtin_amdgcn_wmma_f32_16x16x32_f16(false, a, false, b, (short)0, c,
                                                false, false);
}

static __device__ inline v16h cvt16(float4 a, float4 b, float4 c, float4 d) {
  v16h r;
  r[0]=(_Float16)a.x;  r[1]=(_Float16)a.y;  r[2]=(_Float16)a.z;  r[3]=(_Float16)a.w;
  r[4]=(_Float16)b.x;  r[5]=(_Float16)b.y;  r[6]=(_Float16)b.z;  r[7]=(_Float16)b.w;
  r[8]=(_Float16)c.x;  r[9]=(_Float16)c.y;  r[10]=(_Float16)c.z; r[11]=(_Float16)c.w;
  r[12]=(_Float16)d.x; r[13]=(_Float16)d.y; r[14]=(_Float16)d.z; r[15]=(_Float16)d.w;
  return r;
}

static __device__ inline v16h cat8(v8h lo, v8h hi) {
  return __builtin_shufflevector(lo, hi, 0,1,2,3,4,5,6,7,8,9,10,11,12,13,14,15);
}

// ---------------------------------------------------------------------------
// Kernel 1: Q = x@Wq^T+bq, K = x@Wk^T+bk, V = x@Wv^T+bv  (f32 in, f16 out)
// One wave per (16-row tile, 16-col tile, matrix). 3072 waves = 384 blocks.
// V is stored transposed: Vt[d_global][key]  (d_global = h*16+d)
// ---------------------------------------------------------------------------
__global__ __launch_bounds__(256) void qkv_proj_kernel(
    const float* __restrict__ x,
    const float* __restrict__ Wq, const float* __restrict__ bq,
    const float* __restrict__ Wk, const float* __restrict__ bk,
    const float* __restrict__ Wv, const float* __restrict__ bv,
    _Float16* __restrict__ Qf, _Float16* __restrict__ Kf,
    _Float16* __restrict__ Vt) {
  const int lane = threadIdx.x & 31;
  const int wid  = blockIdx.x * 8 + (threadIdx.x >> 5);   // 0..3071
  const int mat  = wid % 3;            // 0=Q 1=K 2=V
  const int q3   = wid / 3;            // 0..1023
  const int ot   = q3 & 7;             // output col tile
  const int nt   = q3 >> 3;            // row tile 0..127
  const int half = lane >> 4;
  const int n16  = lane & 15;
  const int kbA  = half ? 8 : 0;       // A-operand K base (16-bit A layout)
  const int kbB  = half ? 16 : 0;      // B-operand K base

  const float* W  = (mat == 0) ? Wq : (mat == 1) ? Wk : Wv;
  const float* bb = (mat == 0) ? bq : (mat == 1) ? bk : bv;

  v8f acc = {};
  const float* arow = x + (size_t)(nt * 16 + n16) * HID;   // A row (m = n16)
  const float* brow = W + (size_t)(ot * 16 + n16) * HID;   // B col n: W[o, c] contiguous in c
  #pragma unroll
  for (int c = 0; c < HID; c += 32) {
    const float4* ap0 = (const float4*)(arow + c + kbB);        // K = c+kbB .. +7
    const float4* ap1 = (const float4*)(arow + c + 8 + kbB);    // K = c+8+kbB .. +7
    v16h a = cvt16(ap0[0], ap0[1], ap1[0], ap1[1]);
    const float4* bp = (const float4*)(brow + c + kbB);         // K = c+kbB .. +15
    v16h b = cvt16(bp[0], bp[1], bp[2], bp[3]);
    acc = wmma16(a, b, acc);
  }
  const float bval = bb[ot * 16 + n16];  // per output column
  #pragma unroll
  for (int r = 0; r < 8; ++r) acc[r] += bval;

  if (mat == 2) {
    // transposed store: Vt[d][key]; acc[r] -> key = nt*16 + half*8 + r (contiguous)
    const int d = ot * 16 + n16;
    v8h pk;
    #pragma unroll
    for (int r = 0; r < 8; ++r) pk[r] = (_Float16)acc[r];
    *(v8h*)(Vt + (size_t)d * NSEQ + nt * 16 + half * 8) = pk;
  } else {
    _Float16* dst = (mat == 0) ? Qf : Kf;
    #pragma unroll
    for (int r = 0; r < 8; ++r) {
      const int row = nt * 16 + half * 8 + r;
      dst[(size_t)row * HID + ot * 16 + n16] = (_Float16)acc[r];
    }
  }
}

// ---------------------------------------------------------------------------
// Kernel 2: fused flash attention. One block (8 waves = 8 heads) per 16-query
// tile. The 134MB bias tensor is the HBM-bound stream (~5.8us floor @23.3TB/s):
// it is double-buffered through LDS with cooperative async global->LDS b128
// copies (ASYNCcnt pipeline), fully coalesced (16 rows x 1KB contiguous).
// ---------------------------------------------------------------------------
__global__ __launch_bounds__(256) void flash_attn_kernel(
    const float* __restrict__ Bias,              // [N][N][HEADS] f32
    const unsigned char* __restrict__ kmask,     // [N] bool
    const _Float16* __restrict__ Qf, const _Float16* __restrict__ Kf,
    const _Float16* __restrict__ Vt, _Float16* __restrict__ Ao) {
  // bias tile buffers: per q-row stride 260 f32 (256 payload + 4 pad) keeps the
  // 16B async chunks aligned and limits per-head read bank conflicts to 2-way.
  __shared__ float ldsBias[2][16][260];          // 2 x 16.25 KB
  __shared__ float ldsP[8][16][32];              // per-wave private P staging (16KB)
  const int tid  = threadIdx.x;
  const int lane = tid & 31;
  const int h    = tid >> 5;                     // head = wave id
  const int q0   = blockIdx.x * 16;
  const int half = lane >> 4;
  const int n16  = lane & 15;
  const int kbA  = half ? 8 : 0;
  const float minf = -__builtin_inff();

  // cooperative stage of bias[q0..q0+15][kb..kb+31][0..7]: 1024 x 16B chunks,
  // 4 per thread.  chunk -> row=chunk>>6, c=chunk&63, k=c>>1, part=c&1
  auto issue_bias = [&](int buf, int kb) {
    #pragma unroll
    for (int j = 0; j < 4; ++j) {
      const int chunk = tid + 256 * j;
      const int row = chunk >> 6, c = chunk & 63, k = c >> 1, part = c & 1;
      const float* g =
          Bias + ((size_t)(q0 + row) * NSEQ + kb + k) * NHEADS + part * 4;
      async_copy16(g, &ldsBias[buf][row][k * 8 + part * 4]);
    }
  };

  issue_bias(0, 0);   // prologue: fill buffer 0

  // Q tile (16x16) as A operand, zero-padded to K=32 (dims 16..31 = 0)
  v16h qa = {};
  if (half == 0) {
    const v8h* src = (const v8h*)(Qf + (size_t)(q0 + n16) * HID + h * HDIM);
    qa = cat8(src[0], src[1]);
  }

  v8f acc = {};
  float mrow[8], lrow[8];
  #pragma unroll
  for (int r = 0; r < 8; ++r) { mrow[r] = minf; lrow[r] = 0.f; }

  for (int it = 0; it < NSEQ / 32; ++it) {
    const int kb  = it * 32;
    const int buf = it & 1;

    // software pipeline: kick off next tile's DMA, then drain current tile
    if (kb + 32 < NSEQ) { issue_bias(buf ^ 1, kb + 32); wait_asynccnt<4>(); }
    else                { wait_asynccnt<0>(); }
    __syncthreads();                         // all waves' chunks landed in LDS

    // stream-ahead hint two tiles out (L2 prefetch)
    if (kb + 64 < NSEQ)
      __builtin_prefetch(
          Bias + ((size_t)(q0 + (tid >> 4)) * NSEQ + kb + 64 + (tid & 15) * 2) * NHEADS,
          0, 1);

    // --- S = Q K^T : K-tiles as B operands (lanes 16..31 = zero pad dims) ---
    v16h kt0 = {}, kt1 = {};
    if (half == 0) {
      const v8h* k0 = (const v8h*)(Kf + (size_t)(kb + n16) * HID + h * HDIM);
      const v8h* k1 = (const v8h*)(Kf + (size_t)(kb + 16 + n16) * HID + h * HDIM);
      kt0 = cat8(k0[0], k0[1]);
      kt1 = cat8(k1[0], k1[1]);
    }
    v8f zero = {};
    v8f s0 = wmma16(qa, kt0, zero);   // keys kb..kb+15
    v8f s1 = wmma16(qa, kt1, zero);   // keys kb+16..kb+31

    // --- scale + bias (from LDS tile) + key-padding mask ---
    const int key0 = kb + n16, key1 = kb + 16 + n16;
    const unsigned int mk0 = kmask[key0], mk1 = kmask[key1];
    #pragma unroll
    for (int r = 0; r < 8; ++r) {
      const int qr = half * 8 + r;
      const float b0 = ldsBias[buf][qr][n16 * 8 + h];
      const float b1 = ldsBias[buf][qr][(16 + n16) * 8 + h];
      const float t0 = s0[r] * QKSCALE + b0;
      const float t1 = s1[r] * QKSCALE + b1;
      s0[r] = mk0 ? minf : t0;
      s1[r] = mk1 ? minf : t1;
    }

    // --- online softmax: 16-lane row reductions + accumulator rescale ---
    #pragma unroll
    for (int r = 0; r < 8; ++r) {
      float mb = fmaxf(s0[r], s1[r]);
      #pragma unroll
      for (int off = 1; off < 16; off <<= 1) mb = fmaxf(mb, __shfl_xor(mb, off, 32));
      const float mnew = fmaxf(mrow[r], mb);
      float p0, p1, alpha;
      if (mnew == minf) { p0 = 0.f; p1 = 0.f; alpha = 1.f; }
      else {
        p0 = __expf(s0[r] - mnew);
        p1 = __expf(s1[r] - mnew);
        alpha = (mrow[r] == minf) ? 0.f : __expf(mrow[r] - mnew);
      }
      float rs = p0 + p1;
      #pragma unroll
      for (int off = 1; off < 16; off <<= 1) rs += __shfl_xor(rs, off, 32);
      lrow[r] = lrow[r] * alpha + rs;
      mrow[r] = mnew;
      acc[r] *= alpha;
      s0[r] = p0;
      s1[r] = p1;
    }

    // --- P: C-layout -> LDS -> A-layout (per-wave slice, dscnt-ordered) ---
    #pragma unroll
    for (int r = 0; r < 8; ++r) {
      ldsP[h][half * 8 + r][n16]      = s0[r];
      ldsP[h][half * 8 + r][16 + n16] = s1[r];
    }
    __asm__ volatile("s_wait_dscnt 0" ::: "memory");
    v16h pa;
    {
      const float4* pr  = (const float4*)&ldsP[h][n16][kbA];        // K = kbA..+7
      const float4* pr2 = (const float4*)&ldsP[h][n16][16 + kbA];   // K = 16+kbA..+7
      pa = cvt16(pr[0], pr[1], pr2[0], pr2[1]);
    }
    __asm__ volatile("s_wait_dscnt 0" ::: "memory");

    // --- V tile as B operand: lane col = head dim, elements = 16 keys ---
    v16h vb;
    {
      const int kbB = half ? 8 : 0;
      const v8h* vsrc = (const v8h*)(Vt + (size_t)(h * HDIM + n16) * NSEQ + kb + kbB);
      vb = cat8(vsrc[0], vsrc[2]);
    }
    acc = wmma16(pa, vb, acc);   // rescaled accumulator as C

    __syncthreads();             // all reads of ldsBias[buf] done before reuse
  }

  // finalize: divide by softmax denom, store f16 attention output
  #pragma unroll
  for (int r = 0; r < 8; ++r) {
    const int qg = q0 + half * 8 + r;
    const float o = acc[r] / fmaxf(lrow[r], 1e-30f);
    Ao[(size_t)qg * HID + h * HDIM + n16] = (_Float16)o;
  }
}

// ---------------------------------------------------------------------------
// Kernel 3: out = Ao @ Wo^T + bo  (f32 output)
// ---------------------------------------------------------------------------
__global__ __launch_bounds__(256) void out_proj_kernel(
    const _Float16* __restrict__ Ao, const float* __restrict__ Wo,
    const float* __restrict__ bo, float* __restrict__ out) {
  const int lane = threadIdx.x & 31;
  const int wid  = blockIdx.x * 8 + (threadIdx.x >> 5);   // 0..1023
  const int ot   = wid & 7;
  const int nt   = wid >> 3;
  const int half = lane >> 4;
  const int n16  = lane & 15;
  const int kbA  = half ? 8 : 0;
  const int kbB  = half ? 16 : 0;

  v8f acc = {};
  const _Float16* arow = Ao + (size_t)(nt * 16 + n16) * HID;
  const float*    brow = Wo + (size_t)(ot * 16 + n16) * HID;
  #pragma unroll
  for (int c = 0; c < HID; c += 32) {
    const v8h* ap0 = (const v8h*)(arow + c + kbB);
    const v8h* ap1 = (const v8h*)(arow + c + 8 + kbB);
    v16h a = cat8(ap0[0], ap1[0]);
    const float4* bp = (const float4*)(brow + c + kbB);
    v16h b = cvt16(bp[0], bp[1], bp[2], bp[3]);
    acc = wmma16(a, b, acc);
  }
  const float bias = bo[ot * 16 + n16];
  #pragma unroll
  for (int r = 0; r < 8; ++r) {
    const int row = nt * 16 + half * 8 + r;
    out[(size_t)row * HID + ot * 16 + n16] = acc[r] + bias;
  }
}

// ---------------------------------------------------------------------------
extern "C" void kernel_launch(void* const* d_in, const int* in_sizes, int n_in,
                              void* d_out, int out_size, void* d_ws, size_t ws_size,
                              hipStream_t stream) {
  const float*         x    = (const float*)d_in[0];
  const float*         bias = (const float*)d_in[1];
  const unsigned char* mask = (const unsigned char*)d_in[2];
  const float* Wq = (const float*)d_in[3];
  const float* bq = (const float*)d_in[4];
  const float* Wk = (const float*)d_in[5];
  const float* bk = (const float*)d_in[6];
  const float* Wv = (const float*)d_in[7];
  const float* bv = (const float*)d_in[8];
  const float* Wo = (const float*)d_in[9];
  const float* bo = (const float*)d_in[10];
  float* out = (float*)d_out;

  // workspace: Qf16 | Kf16 | Vt_f16 | AttnOut_f16  (4 * 2048*128 * 2B = 2 MB)
  _Float16* Qf = (_Float16*)d_ws;
  _Float16* Kf = Qf + (size_t)NSEQ * HID;
  _Float16* Vt = Kf + (size_t)NSEQ * HID;
  _Float16* Ao = Vt + (size_t)NSEQ * HID;

  qkv_proj_kernel<<<384, 256, 0, stream>>>(x, Wq, bq, Wk, bk, Wv, bv, Qf, Kf, Vt);
  flash_attn_kernel<<<NSEQ / 16, 256, 0, stream>>>(bias, mask, Qf, Kf, Vt, Ao);
  out_proj_kernel<<<128, 256, 0, stream>>>(Ao, Wo, bo, out);
}
